// Attention_58652073394861
// MI455X (gfx1250) — matrix operand order repaired and checked by reading, not compiled, not run on hardware
//
#include <hip/hip_runtime.h>

// ---------------------------------------------------------------------------
// Multi-head attention for MI455X (gfx1250), wave32 + WMMA f32_16x16x32_f16.
// Async global->LDS staging (ASYNCcnt) for GEMM tiles; NT stores for the
// streaming outputs (probs 537MB, out 34MB) to keep K/V/CTX resident in L2.
// ---------------------------------------------------------------------------

typedef __attribute__((ext_vector_type(16))) _Float16 v16h;
typedef __attribute__((ext_vector_type(8)))  float    v8f;
typedef __attribute__((ext_vector_type(4)))  float    f32x4;
typedef __attribute__((ext_vector_type(2)))  float    f32x2;
typedef __attribute__((ext_vector_type(4)))  unsigned u32x4;
typedef __attribute__((ext_vector_type(4)))  _Float16 f16x4;

#define B_   8
#define S_   1024
#define D_   1024
#define H_   16
#define HD_  64
#define MTOT (B_ * S_)            // 8192 rows

#define USE_ASYNC_LDS 1           // gfx1250 GLOBAL_LOAD_ASYNC_TO_LDS path

union AFrag { v16h v; _Float16 h[16]; unsigned u[8]; };
union BFrag { v16h v; _Float16 h[16]; u32x4 q[2]; };

__device__ __forceinline__ v8f wmma16(v16h a, v16h b, v8f c) {
    // D = A(16x32 f16) * B(32x16 f16) + C(16x16 f32)
    return __builtin_amdgcn_wmma_f32_16x16x32_f16(
        false, a, false, b, (short)0, c, false, false);
}

#if USE_ASYNC_LDS
// Per-lane async 16B copy global -> LDS (tracked by ASYNCcnt, no VGPR data).
__device__ __forceinline__ void async_ld_b128(void* ldsdst, const void* src) {
    unsigned loff = (unsigned)(unsigned long long)ldsdst;  // LDS byte address
    asm volatile("global_load_async_to_lds_b128 %0, %1, off"
                 :: "v"(loff), "v"(src)
                 : "memory");
}
__device__ __forceinline__ void wait_async0() {
    asm volatile("s_wait_asynccnt 0x0" ::: "memory");
}
#endif

// ---------------------------------------------------------------------------
// fp32 -> f16 conversion (vectorized x4)
// ---------------------------------------------------------------------------
__global__ void cvt_kernel(const float* __restrict__ in,
                           _Float16* __restrict__ out, int n) {
    int i = (blockIdx.x * blockDim.x + threadIdx.x) * 4;
    if (i >= n) return;
    f32x4 f = *(const f32x4*)(in + i);
    f16x4 h = __builtin_convertvector(f, f16x4);
    *(f16x4*)(out + i) = h;
}

// fp32 W[k][n] -> f16 WT[n][k]  (D x D)
__global__ void cvtT_kernel(const float* __restrict__ in,
                            _Float16* __restrict__ out) {
    int idx = blockIdx.x * blockDim.x + threadIdx.x;   // < D*D
    int n = idx >> 10, k = idx & (D_ - 1);
    out[idx] = (_Float16)in[k * D_ + n];
}

// ---------------------------------------------------------------------------
// Tiled WMMA GEMM:  Y[M=8192][1024] = A f16 [M][1024] @ (BT f16 [n][k])^T + bias
//   MODE 0: write f16 to [bh][s][hd]   (Q / K layout)
//   MODE 1: write f16 to [bh][hd][s]   (V transposed layout)
//   MODE 2: write fp32 to [m][n]       (final output projection, NT stores)
// Block = 256 threads (8 waves). Block tile 128x128, K-step 32.
// Wave grid 2x4: each wave computes 64x32 (4x2 WMMA accumulators).
// ---------------------------------------------------------------------------
template <int MODE>
__global__ __launch_bounds__(256)
void gemm_kernel(const _Float16* __restrict__ A,
                 const _Float16* __restrict__ BT,
                 const float* __restrict__ bias,
                 void* __restrict__ dst) {
    __shared__ _Float16 sA[128 * 32];
    __shared__ _Float16 sB[128 * 32];

    const int tid  = threadIdx.x;
    const int wid  = tid >> 5;
    const int lane = tid & 31;
    const int lh   = lane & 15;
    const int hi   = lane >> 4;             // 0 or 1
    const int wm   = wid >> 2;              // 0..1
    const int wn   = wid & 3;               // 0..3
    const int m0   = blockIdx.y * 128;
    const int n0   = blockIdx.x * 128;

    v8f acc[4][2];
#pragma unroll
    for (int i = 0; i < 4; ++i)
#pragma unroll
        for (int j = 0; j < 2; ++j)
#pragma unroll
            for (int r = 0; r < 8; ++r) acc[i][j][r] = 0.0f;

    for (int k0 = 0; k0 < D_; k0 += 32) {
        __syncthreads();
        // Stage A tile [128 m][32 k] and BT tile [128 n][32 k].
#pragma unroll
        for (int i = 0; i < 2; ++i) {
            int c   = tid + i * 256;        // 0..511 chunk id
            int row = c >> 2;
            int cc  = c & 3;
#if USE_ASYNC_LDS
            async_ld_b128(&((u32x4*)sA)[row * 4 + cc],
                          &((const u32x4*)A)[(size_t)(m0 + row) * 128 +
                                             (k0 >> 3) + cc]);
            async_ld_b128(&((u32x4*)sB)[row * 4 + cc],
                          &((const u32x4*)BT)[(size_t)(n0 + row) * 128 +
                                              (k0 >> 3) + cc]);
#else
            ((u32x4*)sA)[row * 4 + cc] =
                ((const u32x4*)A)[(size_t)(m0 + row) * 128 + (k0 >> 3) + cc];
            ((u32x4*)sB)[row * 4 + cc] =
                ((const u32x4*)BT)[(size_t)(n0 + row) * 128 + (k0 >> 3) + cc];
#endif
        }
#if USE_ASYNC_LDS
        wait_async0();
#endif
        __syncthreads();

        AFrag af[4];
#pragma unroll
        for (int ms = 0; ms < 4; ++ms) {
            int row = wm * 64 + ms * 16 + lh;
#pragma unroll
            for (int e2 = 0; e2 < 8; ++e2) {
                int kl = 2 * e2 + hi * 16;
                af[ms].u[e2] = *(const unsigned*)&sA[row * 32 + kl];
            }
        }
        BFrag bf[2];
#pragma unroll
        for (int ns = 0; ns < 2; ++ns) {
            int nrow = wn * 32 + ns * 16 + lh;
            const u32x4* p = (const u32x4*)&sB[nrow * 32 + hi * 16];
            bf[ns].q[0] = p[0];
            bf[ns].q[1] = p[1];
        }
#pragma unroll
        for (int ms = 0; ms < 4; ++ms)
#pragma unroll
            for (int ns = 0; ns < 2; ++ns)
                acc[ms][ns] = wmma16(af[ms].v, bf[ns].v, acc[ms][ns]);
    }

    // Epilogue: bias + layout-specific store.
#pragma unroll
    for (int ns = 0; ns < 2; ++ns) {
        int nc = n0 + wn * 32 + ns * 16 + lh;
        float bv = bias[nc];
#pragma unroll
        for (int ms = 0; ms < 4; ++ms) {
#pragma unroll
            for (int r = 0; r < 8; ++r) {
                int   mr  = m0 + wm * 64 + ms * 16 + r + hi * 8;
                float val = acc[ms][ns][r] + bv;
                if (MODE == 2) {
                    // Streaming fp32 result: bypass far caches.
                    __builtin_nontemporal_store(
                        val, &((float*)dst)[(size_t)mr * D_ + nc]);
                } else {
                    int b = mr >> 10, s = mr & (S_ - 1);
                    int h = nc >> 6,  hd = nc & (HD_ - 1);
                    size_t idx = (MODE == 0)
                        ? ((size_t)(b * H_ + h) * S_ + s) * HD_ + hd
                        : ((size_t)(b * H_ + h) * HD_ + hd) * S_ + s;
                    ((_Float16*)dst)[idx] = (_Float16)val;
                }
            }
        }
    }
}

// ---------------------------------------------------------------------------
// Fused attention: scores = QK^T/8, softmax (probs -> global fp32), ctx = P@V.
// Grid: (S/64 q-tiles, B*H). Block: 128 threads = 4 waves, each wave owns
// 16 q-rows. Dynamic LDS: 64 x 1024 fp32 score tile (256 KB, <= 320 KB/WGP).
// ---------------------------------------------------------------------------
__global__ __launch_bounds__(128)
void attn_kernel(const _Float16* __restrict__ Q,    // [bh][s][64]
                 const _Float16* __restrict__ K,    // [bh][s][64]
                 const _Float16* __restrict__ VT,   // [bh][64][s]
                 float* __restrict__ probs,         // [bh][s][s]
                 _Float16* __restrict__ ctx) {      // [b][s][1024]
    extern __shared__ float sc[];                   // 64 rows x 1024 cols

    const int tid  = threadIdx.x;
    const int wid  = tid >> 5;
    const int lane = tid & 31;
    const int lh   = lane & 15;
    const int hi   = lane >> 4;
    const int bh   = blockIdx.y;                    // 0..127
    const int q0   = blockIdx.x * 64;
    const int b    = bh >> 4, h = bh & 15;

    float* myrows = sc + wid * 16 * S_;             // wave-private 16 x 1024

    // ---- Q fragments (A operand, K-dim = HD = 64 -> 2 k-steps) ----
    AFrag qf[2];
    {
        int qrow = q0 + wid * 16 + lh;
        const unsigned* qp =
            (const unsigned*)(Q + ((size_t)bh * S_ + qrow) * HD_);
#pragma unroll
        for (int ks = 0; ks < 2; ++ks)
#pragma unroll
            for (int e2 = 0; e2 < 8; ++e2) {
                int kl = ks * 32 + 2 * e2 + hi * 16;
                qf[ks].u[e2] = qp[kl >> 1];
            }
    }

    // ---- Phase 1: scores tile (16 x 1024 per wave) via WMMA ----
    for (int nt = 0; nt < 64; ++nt) {
        v8f a;
#pragma unroll
        for (int r = 0; r < 8; ++r) a[r] = 0.0f;
        int sp = nt * 16 + lh;                       // key row (column of S)
        const u32x4* kp = (const u32x4*)(K + ((size_t)bh * S_ + sp) * HD_);
#pragma unroll
        for (int ks = 0; ks < 2; ++ks) {
            BFrag bf;
            bf.q[0] = kp[ks * 4 + hi * 2 + 0];
            bf.q[1] = kp[ks * 4 + hi * 2 + 1];
            a = wmma16(qf[ks].v, bf.v, a);
        }
#pragma unroll
        for (int r = 0; r < 8; ++r)
            myrows[(r + hi * 8) * S_ + nt * 16 + lh] = a[r] * 0.125f;
    }

    // ---- Phase 2: row softmax (per-wave, 16 rows), write probs ----
    for (int r = 0; r < 16; ++r) {
        float* rowp = myrows + r * S_;
        const f32x4* r4 = (const f32x4*)rowp;
        f32x4 v[8];
        float m = -3.0e38f;
#pragma unroll
        for (int c = 0; c < 8; ++c) {
            v[c] = r4[c * 32 + lane];
            m = fmaxf(m, fmaxf(fmaxf(v[c].x, v[c].y), fmaxf(v[c].z, v[c].w)));
        }
#pragma unroll
        for (int off = 16; off > 0; off >>= 1)
            m = fmaxf(m, __shfl_xor(m, off, 32));
        float sum = 0.0f;
#pragma unroll
        for (int c = 0; c < 8; ++c) {
            v[c].x = __expf(v[c].x - m);
            v[c].y = __expf(v[c].y - m);
            v[c].z = __expf(v[c].z - m);
            v[c].w = __expf(v[c].w - m);
            sum += v[c].x + v[c].y + v[c].z + v[c].w;
        }
#pragma unroll
        for (int off = 16; off > 0; off >>= 1)
            sum += __shfl_xor(sum, off, 32);
        float inv = 1.0f / sum;
        size_t pbase = ((size_t)bh * S_ + (q0 + wid * 16 + r)) * S_;
#pragma unroll
        for (int c = 0; c < 8; ++c) {
            f32x4 p = v[c] * inv;
            ((f32x4*)rowp)[c * 32 + lane] = p;                       // LDS
            // Streaming 537MB probs tensor: non-temporal, don't evict K/V.
            __builtin_nontemporal_store(
                p, (f32x4*)(probs + pbase + (size_t)(c * 32 + lane) * 4));
        }
    }

    // ---- Phase 3: ctx = P(16x1024) @ V(1024x64) via WMMA ----
    v8f acc[4];
#pragma unroll
    for (int i = 0; i < 4; ++i)
#pragma unroll
        for (int r = 0; r < 8; ++r) acc[i][r] = 0.0f;

    for (int ks = 0; ks < 32; ++ks) {
        AFrag af;
        const float* rp = myrows + lh * S_ + ks * 32;
#pragma unroll
        for (int e2 = 0; e2 < 8; ++e2) {
            int kl = 2 * e2 + hi * 16;
            f32x2 f = *(const f32x2*)(rp + kl);
            af.h[2 * e2]     = (_Float16)f.x;
            af.h[2 * e2 + 1] = (_Float16)f.y;
        }
#pragma unroll
        for (int nt = 0; nt < 4; ++nt) {
            int n = nt * 16 + lh;                    // hd column
            const u32x4* vp = (const u32x4*)(
                VT + ((size_t)bh * HD_ + n) * S_ + ks * 32 + hi * 16);
            BFrag bf;
            bf.q[0] = vp[0];
            bf.q[1] = vp[1];
            acc[nt] = wmma16(af.v, bf.v, acc[nt]);
        }
    }
#pragma unroll
    for (int nt = 0; nt < 4; ++nt) {
        int hd = nt * 16 + lh;
#pragma unroll
        for (int r = 0; r < 8; ++r) {
            int srow = q0 + wid * 16 + r + hi * 8;
            ctx[((size_t)b * S_ + srow) * D_ + h * HD_ + hd] =
                (_Float16)acc[nt][r];
        }
    }
}

// ---------------------------------------------------------------------------
// Launcher
// ---------------------------------------------------------------------------
extern "C" void kernel_launch(void* const* d_in, const int* in_sizes, int n_in,
                              void* d_out, int out_size, void* d_ws,
                              size_t ws_size, hipStream_t stream) {
    const float* hs = (const float*)d_in[0];
    const float* Wq = (const float*)d_in[1];
    const float* bq = (const float*)d_in[2];
    const float* Wk = (const float*)d_in[3];
    const float* bk = (const float*)d_in[4];
    const float* Wv = (const float*)d_in[5];
    const float* bv = (const float*)d_in[6];
    const float* Wo = (const float*)d_in[7];
    const float* bo = (const float*)d_in[8];

    float* out   = (float*)d_out;                       // [B,S,D]
    float* probs = out + (size_t)B_ * S_ * D_;          // [B,H,S,S]

    // Workspace layout (f16), total ~88 MB.
    char* ws = (char*)d_ws;
    const size_t SZ_X   = (size_t)MTOT * D_ * 2;        // 16 MB
    const size_t SZ_W   = (size_t)D_ * D_ * 2;          // 2 MB
    const size_t SZ_QKV = (size_t)B_ * H_ * S_ * HD_ * 2;  // 16 MB
    _Float16* X16  = (_Float16*)(ws);
    _Float16* WqT  = (_Float16*)(ws + SZ_X);
    _Float16* WkT  = (_Float16*)(ws + SZ_X + SZ_W);
    _Float16* WvT  = (_Float16*)(ws + SZ_X + 2 * SZ_W);
    _Float16* WoT  = (_Float16*)(ws + SZ_X + 3 * SZ_W);
    _Float16* Q16  = (_Float16*)(ws + SZ_X + 4 * SZ_W);
    _Float16* K16  = (_Float16*)(ws + SZ_X + 4 * SZ_W + SZ_QKV);
    _Float16* VT16 = (_Float16*)(ws + SZ_X + 4 * SZ_W + 2 * SZ_QKV);
    _Float16* CTX  = (_Float16*)(ws + SZ_X + 4 * SZ_W + 3 * SZ_QKV);

    // 1) Precision conversion (X straight, weights transposed).
    {
        int n = MTOT * D_;
        cvt_kernel<<<n / (256 * 4), 256, 0, stream>>>(hs, X16, n);
        int gw = (D_ * D_) / 256;
        cvtT_kernel<<<gw, 256, 0, stream>>>(Wq, WqT);
        cvtT_kernel<<<gw, 256, 0, stream>>>(Wk, WkT);
        cvtT_kernel<<<gw, 256, 0, stream>>>(Wv, WvT);
        cvtT_kernel<<<gw, 256, 0, stream>>>(Wo, WoT);
    }

    // 2) QKV projections (WMMA GEMM).
    dim3 ggrid(D_ / 128, MTOT / 128);                   // (8, 64)
    gemm_kernel<0><<<ggrid, 256, 0, stream>>>(X16, WqT, bq, (void*)Q16);
    gemm_kernel<0><<<ggrid, 256, 0, stream>>>(X16, WkT, bk, (void*)K16);
    gemm_kernel<1><<<ggrid, 256, 0, stream>>>(X16, WvT, bv, (void*)VT16);

    // 3) Fused scores + softmax(+probs out) + ctx.
    const int attn_lds = 64 * S_ * (int)sizeof(float);  // 256 KB
    (void)hipFuncSetAttribute((const void*)attn_kernel,
                              hipFuncAttributeMaxDynamicSharedMemorySize,
                              attn_lds);
    attn_kernel<<<dim3(S_ / 64, B_ * H_), 128, attn_lds, stream>>>(
        Q16, K16, VT16, probs, CTX);

    // 4) Output projection (fp32 result + bias).
    gemm_kernel<2><<<ggrid, 256, 0, stream>>>(CTX, WoT, bo, (void*)out);
}
